// POSTagger_14534169330052
// MI455X (gfx1250) — matrix operand order repaired and checked by reading, not compiled, not run on hardware
//
#include <hip/hip_runtime.h>
#include <hip/hip_bf16.h>
#include <math.h>

typedef __attribute__((ext_vector_type(16))) _Float16 v16h;
typedef __attribute__((ext_vector_type(8)))  float    v8f;

#define S_WORDS 16384

// ---- fast transcendentals: single TRANS ops, no libm branches in WMMA loops ----
__device__ __forceinline__ float fast_exp2(float x) {
#if __has_builtin(__builtin_amdgcn_exp2f)
  return __builtin_amdgcn_exp2f(x);
#else
  return exp2f(x);
#endif
}
__device__ __forceinline__ float fast_rcp(float x) {
#if __has_builtin(__builtin_amdgcn_rcpf)
  return __builtin_amdgcn_rcpf(x);
#else
  return 1.0f / x;
#endif
}
__device__ __forceinline__ float fast_log2(float x) {
#if __has_builtin(__builtin_amdgcn_logf)
  return __builtin_amdgcn_logf(x);
#else
  return log2f(x);
#endif
}
__device__ __forceinline__ float sigf(float x) {
  // 1/(1+2^(-x*log2e))
  return fast_rcp(1.0f + fast_exp2(-1.4426950408889634f * x));
}
__device__ __forceinline__ float fast_tanh(float x) {
#if __has_builtin(__builtin_amdgcn_tanh_f32)
  return __builtin_amdgcn_tanh_f32(x);
#elif __has_builtin(__builtin_amdgcn_tanhf)
  return __builtin_amdgcn_tanhf(x);
#else
  // tanh(x) = 1 - 2/(2^(2x*log2e)+1)
  float e = fast_exp2(2.8853900817779268f * x);
  return 1.0f - 2.0f * fast_rcp(e + 1.0f);
#endif
}

__device__ __forceinline__ void wait_asynccnt0() {
#if __has_builtin(__builtin_amdgcn_s_wait_asynccnt)
  __builtin_amdgcn_s_wait_asynccnt(0);
#else
  asm volatile("s_wait_asynccnt 0x0" ::: "memory");
#endif
}

// WMMA f32 = f16[16x32] * f16[32x16] + f32[16x16]
__device__ __forceinline__ v8f wmma16(v16h a, v16h b, v8f c) {
  return __builtin_amdgcn_wmma_f32_16x16x32_f16(false, a, false, b, (short)0, c, false, false);
}

// A-fragment (16x32, f16) from row-major [M][ld] storage at column base k0.
__device__ __forceinline__ v16h load_a(const _Float16* base, int ld, int k0, int lane) {
  const int m = lane & 15;
  const int hs8 = (lane >> 4) * 8;
  const _Float16* p = base + m * ld + k0 + hs8;
  v16h a;
#pragma unroll
  for (int j = 0; j < 8; ++j) { a[j] = p[j]; a[j + 8] = p[16 + j]; }
  return a;
}

// B-fragment (32x16, f16) from row-major [K][ldb] storage at (k0, n0).
__device__ __forceinline__ v16h load_b(const _Float16* B, int ldb, int k0, int n0, int lane) {
  const int n = lane & 15;
  const int ks = (lane >> 4) * 8;
  const _Float16* p = B + (size_t)(k0 + ks) * ldb + n0 + n;
  v16h b;
#pragma unroll
  for (int j = 0; j < 16; ++j) b[j] = p[(size_t)(j < 8 ? j : j + 8) * ldb];
  return b;
}

// ---------------- weight prep (f32 -> f16, transposes, bias fusion) ----------------
__global__ void k_prep_char(const float* __restrict__ Wih, const float* __restrict__ Whh,
                            const float* __restrict__ bih, const float* __restrict__ bhh,
                            _Float16* __restrict__ WihT, _Float16* __restrict__ WhhT,
                            float* __restrict__ bias) {
  int i = blockIdx.x * blockDim.x + threadIdx.x;
  if (i < 512 * 64)  { int j = i / 64,  k = i & 63;  WihT[k * 512 + j] = (_Float16)Wih[i]; }
  if (i < 512 * 128) { int j = i / 128, k = i & 127; WhhT[k * 512 + j] = (_Float16)Whh[i]; }
  if (i < 512) bias[i] = bih[i] + bhh[i];
}

__global__ void k_prep_word(const float* __restrict__ Wih, const float* __restrict__ Whh,
                            const float* __restrict__ bih, const float* __restrict__ bhh,
                            _Float16* __restrict__ WihT, _Float16* __restrict__ Whh16,
                            float* __restrict__ bias, float* __restrict__ hbuf,
                            int* __restrict__ syncc) {
  int i = blockIdx.x * blockDim.x + threadIdx.x;
  if (i < 1024 * 128) { int j = i / 128, k = i & 127; WihT[k * 1024 + j] = (_Float16)Wih[i]; }
  if (i < 1024 * 256) Whh16[i] = (_Float16)Whh[i];
  if (i < 1024) bias[i] = bih[i] + bhh[i];
  if (i < 256)  hbuf[i] = 0.0f;
  if (i == 0)   *syncc = 0;
}

// ---------------- char LSTM: 16 words per WG, 8 waves, WMMA gates ----------------
__global__ __launch_bounds__(256) void k_char_lstm(
    const int* __restrict__ char_ids, const int* __restrict__ char_lens,
    const float* __restrict__ char_emb,
    const _Float16* __restrict__ WihT /*[64][512]*/,
    const _Float16* __restrict__ WhhT /*[128][512]*/,
    const float* __restrict__ bias /*[512]*/,
    float* __restrict__ out_hidden /* [S][384], char cols at +256 */) {
  constexpr int XLD = 72, HLD = 136;
  __shared__ _Float16 xs[16 * XLD];
  __shared__ _Float16 hsm[16 * HLD];
  __shared__ float bsh[512];
  __shared__ int lens[16];

  const int tid = threadIdx.x, lane = tid & 31, wave = tid >> 5;
  const int word0 = blockIdx.x * 16;

  for (int i = tid; i < 512; i += 256) bsh[i] = bias[i];
  if (tid < 16) lens[tid] = char_lens[word0 + tid];
  for (int i = tid; i < 16 * HLD; i += 256) hsm[i] = (_Float16)0.0f;
  __syncthreads();

  const int nh = wave;  // hidden-column tile [nh*16, nh*16+16)
  // Hoist invariant weight B-fragments into registers (reused 16 steps).
  v16h bi[4][2], bh[4][4];
#pragma unroll
  for (int q = 0; q < 4; ++q) {
#pragma unroll
    for (int kc = 0; kc < 2; ++kc) bi[q][kc] = load_b(WihT, 512, kc * 32, q * 128 + nh * 16, lane);
#pragma unroll
    for (int kc = 0; kc < 4; ++kc) bh[q][kc] = load_b(WhhT, 512, kc * 32, q * 128 + nh * 16, lane);
  }

  v8f c = {0.f, 0.f, 0.f, 0.f, 0.f, 0.f, 0.f, 0.f};
  const int n = lane & 15, mo = (lane >> 4) * 8;

  for (int t = 0; t < 16; ++t) {
    // gather + relu char embeddings for this timestep: 16 words x 64 dims
    for (int i = tid; i < 16 * 64; i += 256) {
      int m = i >> 6, k = i & 63;
      int cid = char_ids[(size_t)(word0 + m) * 16 + t];
      float v = char_emb[(size_t)cid * 64 + k];
      xs[m * XLD + k] = (_Float16)(v > 0.f ? v : 0.f);
    }
    __syncthreads();

    v8f acc[4];
#pragma unroll
    for (int q = 0; q < 4; ++q) {
      float bv = bsh[q * 128 + nh * 16 + n];
      v8f tmp = {bv, bv, bv, bv, bv, bv, bv, bv};
      acc[q] = tmp;
    }
    // input contribution (K=64)
    v16h a0 = load_a(xs, XLD, 0, lane);
    v16h a1 = load_a(xs, XLD, 32, lane);
#pragma unroll
    for (int q = 0; q < 4; ++q) {
      acc[q] = wmma16(a0, bi[q][0], acc[q]);
      acc[q] = wmma16(a1, bi[q][1], acc[q]);
    }
    // recurrent contribution (K=128)
#pragma unroll
    for (int kc = 0; kc < 4; ++kc) {
      v16h ah = load_a(hsm, HLD, kc * 32, lane);
#pragma unroll
      for (int q = 0; q < 4; ++q) acc[q] = wmma16(ah, bh[q][kc], acc[q]);
    }
    // elementwise LSTM cell (pure TRANS-op nonlinearities, no branches)
    v8f hv;
#pragma unroll
    for (int r = 0; r < 8; ++r) {
      float ig = sigf(acc[0][r]), fg = sigf(acc[1][r]);
      float gg = fast_tanh(acc[2][r]), og = sigf(acc[3][r]);
      float cc = fg * c[r] + ig * gg;
      c[r] = cc;
      hv[r] = og * fast_tanh(cc);
    }
    __syncthreads();
#pragma unroll
    for (int r = 0; r < 8; ++r) {
      int m = mo + r;  // C layout: VGPR r -> row r (lanes 0-15) / 8+r (lanes 16-31)
      hsm[m * HLD + nh * 16 + n] = (_Float16)hv[r];
      if (lens[m] - 1 == t)
        out_hidden[(size_t)(word0 + m) * 384 + 256 + nh * 16 + n] = hv[r];
    }
    __syncthreads();
  }
}

// ---------------- word input-gate GEMM: Gx = relu(emb)@WihT + bias ----------------
__global__ __launch_bounds__(256) void k_word_ingemm(
    const int* __restrict__ sentence, const float* __restrict__ word_emb,
    const _Float16* __restrict__ WihT /*[128][1024]*/,
    const float* __restrict__ bias /*[1024]*/, float* __restrict__ Gx /*[S][1024]*/) {
  constexpr int XLD = 136;
  __shared__ _Float16 xs[16 * XLD];
  const int tid = threadIdx.x, lane = tid & 31, wave = tid >> 5;
  const int s0 = blockIdx.x * 16;

  for (int i = tid; i < 16 * 128; i += 256) {
    int m = i >> 7, k = i & 127;
    float v = word_emb[(size_t)sentence[s0 + m] * 128 + k];
    xs[m * XLD + k] = (_Float16)(v > 0.f ? v : 0.f);
  }
  __syncthreads();

  v16h a[4];
#pragma unroll
  for (int kc = 0; kc < 4; ++kc) a[kc] = load_a(xs, XLD, kc * 32, lane);

  const int n = lane & 15, mo = (lane >> 4) * 8;
#pragma unroll
  for (int nt = 0; nt < 8; ++nt) {
    int n0 = wave * 128 + nt * 16;
    float bv = bias[n0 + n];
    v8f acc = {bv, bv, bv, bv, bv, bv, bv, bv};
#pragma unroll
    for (int kc = 0; kc < 4; ++kc) {
      v16h b = load_b(WihT, 1024, kc * 32, n0, lane);
      acc = wmma16(a[kc], b, acc);
    }
#pragma unroll
    for (int r = 0; r < 8; ++r)
      Gx[(size_t)(s0 + mo + r) * 1024 + n0 + n] = acc[r];
  }
}

// ------------- word LSTM recurrence: 4 persistent WGs, LDS-resident Whh slice -------------
__global__ __launch_bounds__(256) void k_word_seq(
    const _Float16* __restrict__ Whh16 /*[1024][256]*/, const float* __restrict__ Gx,
    float* __restrict__ hbuf /*[256]*/, int* __restrict__ syncc,
    float* __restrict__ out_hidden /* [S][384], word cols 0..255 */) {
  constexpr int WLD = 258;  // stride 258 halves -> 129 dwords -> bank stride 1, conflict-free
  __shared__ _Float16 Wl[256 * WLD];
  __shared__ float gsh[256];
  __shared__ float hloc[256];
  __shared__ float csh[64];

  const int tid = threadIdx.x;
  const int wg = blockIdx.x;            // 0..3, owns hidden units [wg*64, wg*64+64)
  const int q = tid >> 6, u = tid & 63; // gate type, unit-in-slice
  const int grow = q * 256 + wg * 64 + u;

  for (int k = 0; k < 256; ++k) Wl[tid * WLD + k] = Whh16[(size_t)grow * 256 + k];
  if (tid < 256) hloc[tid] = 0.f;
  if (tid < 64) csh[tid] = 0.f;
  __syncthreads();

  for (int t = 0; t < S_WORDS; ++t) {
    float acc = Gx[(size_t)t * 1024 + grow];
    if (t + 1 < S_WORDS)  // hide L2 latency of next step's input gates
      __builtin_prefetch(&Gx[(size_t)(t + 1) * 1024 + grow], 0, 0);
    const _Float16* wr = &Wl[tid * WLD];
#pragma unroll 8
    for (int k = 0; k < 256; ++k) acc += (float)wr[k] * hloc[k];
    gsh[q * 64 + u] = acc;
    __syncthreads();

    if (tid < 64) {
      float ig = sigf(gsh[tid]), fg = sigf(gsh[64 + tid]);
      float gg = fast_tanh(gsh[128 + tid]), og = sigf(gsh[192 + tid]);
      float cc = fg * csh[tid] + ig * gg;
      csh[tid] = cc;
      float hh = og * fast_tanh(cc);
      __hip_atomic_store(&hbuf[wg * 64 + tid], hh, __ATOMIC_RELAXED, __HIP_MEMORY_SCOPE_AGENT);
      out_hidden[(size_t)t * 384 + wg * 64 + tid] = hh;
    }
    __syncthreads();
    __threadfence();
    if (tid == 0) {
      __hip_atomic_fetch_add(syncc, 1, __ATOMIC_ACQ_REL, __HIP_MEMORY_SCOPE_AGENT);
      while (__hip_atomic_load(syncc, __ATOMIC_ACQUIRE, __HIP_MEMORY_SCOPE_AGENT) < 4 * (t + 1))
        __builtin_amdgcn_s_sleep(1);
    }
    __syncthreads();
    if (tid < 256) hloc[tid] = __hip_atomic_load(&hbuf[tid], __ATOMIC_RELAXED, __HIP_MEMORY_SCOPE_AGENT);
    __syncthreads();
  }
}

// ---------------- final linear + log_softmax: one wave per word ----------------
__global__ __launch_bounds__(256) void k_linear_lsm(
    const float* __restrict__ hidden /*[S][384]*/, const float* __restrict__ W /*[50][384]*/,
    const float* __restrict__ b /*[50]*/, float* __restrict__ outp /*[50][S]*/) {
  __shared__ float Ws[50 * 384];
  __shared__ float bs[50];
  const int tid = threadIdx.x, lane = tid & 31, wave = tid >> 5;

  // Stage lin_W into LDS via the gfx1250 async DMA-to-LDS path (ASYNCcnt-tracked):
  // 50*384 floats = 4800 x 16B transfers across 256 lanes.
  {
    unsigned ldsbase = (unsigned)(uintptr_t)(&Ws[0]);  // low 32 bits = LDS offset
    for (int i = tid; i < 4800; i += 256) {
      unsigned lds = ldsbase + (unsigned)i * 16u;
      unsigned long long ga = (unsigned long long)(uintptr_t)W + (unsigned long long)i * 16ull;
      asm volatile("global_load_async_to_lds_b128 %0, %1, off"
                   :: "v"(lds), "v"(ga) : "memory");
    }
    wait_asynccnt0();
  }
  if (tid < 50) bs[tid] = b[tid];
  __syncthreads();

  const int s = blockIdx.x * 8 + wave;
  const int t0 = lane, t1 = lane + 32;
  float a0 = bs[t0];
  float a1 = (t1 < 50) ? bs[t1] : 0.f;
  const float* hrow = hidden + (size_t)s * 384;
  for (int k = 0; k < 384; ++k) {
    float hv = hrow[k];
    a0 += hv * Ws[t0 * 384 + k];
    if (t1 < 50) a1 += hv * Ws[t1 * 384 + k];
  }
  float m = a0;
  if (t1 < 50 && a1 > m) m = a1;
#pragma unroll
  for (int off = 16; off > 0; off >>= 1) { float o = __shfl_xor(m, off, 32); if (o > m) m = o; }
  constexpr float LOG2E = 1.4426950408889634f, LN2 = 0.6931471805599453f;
  float e = fast_exp2((a0 - m) * LOG2E) + ((t1 < 50) ? fast_exp2((a1 - m) * LOG2E) : 0.f);
#pragma unroll
  for (int off = 16; off > 0; off >>= 1) e += __shfl_xor(e, off, 32);
  float lse = m + fast_log2(e) * LN2;
  outp[(size_t)t0 * S_WORDS + s] = a0 - lse;
  if (t1 < 50) outp[(size_t)t1 * S_WORDS + s] = a1 - lse;
}

extern "C" void kernel_launch(void* const* d_in, const int* in_sizes, int n_in,
                              void* d_out, int out_size, void* d_ws, size_t ws_size,
                              hipStream_t stream) {
  (void)in_sizes; (void)n_in; (void)out_size; (void)ws_size;
  const int*   sentence  = (const int*)d_in[0];
  const int*   char_ids  = (const int*)d_in[1];
  const int*   char_lens = (const int*)d_in[2];
  const float* char_emb  = (const float*)d_in[3];
  const float* word_emb  = (const float*)d_in[4];
  const float* char_Wih  = (const float*)d_in[5];
  const float* char_Whh  = (const float*)d_in[6];
  const float* char_bih  = (const float*)d_in[7];
  const float* char_bhh  = (const float*)d_in[8];
  const float* word_Wih  = (const float*)d_in[9];
  const float* word_Whh  = (const float*)d_in[10];
  const float* word_bih  = (const float*)d_in[11];
  const float* word_bhh  = (const float*)d_in[12];
  const float* lin_W     = (const float*)d_in[13];
  const float* lin_b     = (const float*)d_in[14];

  float* outp   = (float*)d_out;                     // logp [50][16384]
  float* hidden = outp + (size_t)50 * S_WORDS;       // hidden [16384][384]

  size_t o = 0;
  char* wsb = (char*)d_ws;
  auto take = [&](size_t bytes) -> void* {
    void* p = wsb + o;
    o += (bytes + 255) & ~(size_t)255;
    return p;
  };
  _Float16* cWihT  = (_Float16*)take((size_t)64 * 512 * 2);
  _Float16* cWhhT  = (_Float16*)take((size_t)128 * 512 * 2);
  float*    cBias  = (float*)take(512 * 4);
  _Float16* wWihT  = (_Float16*)take((size_t)128 * 1024 * 2);
  _Float16* wWhh16 = (_Float16*)take((size_t)1024 * 256 * 2);
  float*    wBias  = (float*)take(1024 * 4);
  float*    hbuf   = (float*)take(256 * 4);
  int*      syncc  = (int*)take(256);
  float*    Gx     = (float*)take((size_t)S_WORDS * 1024 * 4);

  k_prep_char<<<256, 256, 0, stream>>>(char_Wih, char_Whh, char_bih, char_bhh,
                                       cWihT, cWhhT, cBias);
  k_prep_word<<<1024, 256, 0, stream>>>(word_Wih, word_Whh, word_bih, word_bhh,
                                        wWihT, wWhh16, wBias, hbuf, syncc);
  k_char_lstm<<<S_WORDS / 16, 256, 0, stream>>>(char_ids, char_lens, char_emb,
                                                cWihT, cWhhT, cBias, hidden);
  k_word_ingemm<<<S_WORDS / 16, 256, 0, stream>>>(sentence, word_emb, wWihT, wBias, Gx);
  k_word_seq<<<4, 256, 0, stream>>>(wWhh16, Gx, hbuf, syncc, hidden);
  k_linear_lsm<<<S_WORDS / 8, 256, 0, stream>>>(hidden, lin_W, lin_b, outp);
}
